// Attention_42425686950371
// MI455X (gfx1250) — matrix operand order repaired and checked by reading, not compiled, not run on hardware
//
#include <hip/hip_runtime.h>
#include <math.h>

typedef __attribute__((ext_vector_type(16))) _Float16 v16h;
typedef __attribute__((ext_vector_type(8)))  _Float16 v8h;
typedef __attribute__((ext_vector_type(8)))  float    v8f;

#define DD 256
#define TT 4096
#define BB 4

union HV { v16h v; v8h h[2]; };

// ---------------- f32 -> f16 convert ----------------
__global__ void cvt_f32_f16(const float* __restrict__ src, _Float16* __restrict__ dst, int n) {
    int i = blockIdx.x * blockDim.x + threadIdx.x;
    if (i < n) dst[i] = (_Float16)src[i];
}

// ---------------- QKV projection: dst = (x @ W^T + b) * scale ----------------
// grid: (TT/128, BB), block: 256 (8 waves). Wave w computes rows [mbase, mbase+16) x all 256 cols.
template <bool TRANSPOSED>
__global__ __launch_bounds__(256) void qkv_gemm(
    const _Float16* __restrict__ Xh,   // [B,T,D] row-major f16
    const _Float16* __restrict__ Wh,   // [D,D]  row-major (n,k) f16
    const float*    __restrict__ bias, // [D] f32
    _Float16* __restrict__ dst,        // [B,T,D] or transposed [B,D,T]
    float scale)
{
    __shared__ __align__(128) _Float16 Sb[8][16][16];   // per-wave store-staging tiles

    const int lane  = threadIdx.x & 31;
    const int wave  = threadIdx.x >> 5;
    const int b     = blockIdx.y;
    const int mbase = blockIdx.x * 128 + wave * 16;
    const int l15   = lane & 15;
    const int hi    = lane >> 4;

    // Preload A fragments (x rows), full K = 256 in 8 chunks of 32.
    // f16 A layout: lanes 0-15 hold K = 0..15, lanes 16-31 hold K = 16..31 (contiguous, matches B).
    const _Float16* xrow = Xh + ((size_t)b * TT + mbase + l15) * DD;
    HV a[8];
#pragma unroll
    for (int kk = 0; kk < 8; ++kk) {
        const _Float16* p = xrow + kk * 32 + hi * 16;
        a[kk].h[0] = *(const v8h*)(p);
        a[kk].h[1] = *(const v8h*)(p + 8);
    }

#pragma unroll
    for (int nt = 0; nt < 16; ++nt) {
        v8f acc = {};
        // B layout: lane n<16 holds column n, K=0..15 contiguous; lanes 16-31 K=16..31.
        // B[k,n] = W[n,k] -> contiguous along W row. 32B aligned.
        const _Float16* wr = Wh + (size_t)(nt * 16 + l15) * DD + hi * 16;
#pragma unroll
        for (int kk = 0; kk < 8; ++kk) {
            v16h bf = *(const v16h*)(wr + kk * 32);
            acc = __builtin_amdgcn_wmma_f32_16x16x32_f16(false, a[kk].v, false, bf,
                                                         (short)0, acc, false, false);
        }
        const int col = nt * 16 + l15;
        const float bvv = bias[col];

        if (TRANSPOSED) {
            // C layout: slot r = row (mbase + hi*8 + r), fixed col -> 8 consecutive
            // t-positions of one V^T row: pack into a single 16B store.
            v8h hv;
#pragma unroll
            for (int r = 0; r < 8; ++r) hv[r] = (_Float16)((acc[r] + bvv) * scale);
            *(v8h*)(dst + ((size_t)b * DD + col) * TT + mbase + hi * 8) = hv;
        } else {
            // Stage tile through LDS -> coalesced b128 stores.
#pragma unroll
            for (int r = 0; r < 8; ++r)
                Sb[wave][r + hi * 8][l15] = (_Float16)((acc[r] + bvv) * scale);
            asm volatile("s_wait_dscnt 0" ::: "memory");
            v8h hv = *(const v8h*)&Sb[wave][lane >> 1][(lane & 1) * 8];
            const int row = mbase + (lane >> 1);
            const int c0  = nt * 16 + (lane & 1) * 8;
            *(v8h*)(dst + ((size_t)b * TT + row) * DD + c0) = hv;
            asm volatile("" ::: "memory");   // keep next tile's DS stores below this load
        }
    }
}

// ---------------- causal flash attention (transposed formulation) ----------------
// S^T = K Q^T  (A = K rows, B = Q rows)   -> C layout: row = key s, col = query q
// O^T = V^T P^T (A = V^T rows, B = P^T)   -> C layout: row = d,     col = query q
// grid: (TT/64, BB), block 128 (4 independent waves, 16 query rows each). No LDS.
__global__ __launch_bounds__(128) void flash_attn(
    const _Float16* __restrict__ Qh,  // [B,T,D] pre-scaled by 1/sqrt(D)
    const _Float16* __restrict__ Kh,  // [B,T,D]
    const _Float16* __restrict__ Vt,  // [B,D,T] (V transposed)
    float* __restrict__ out)          // [1,B,T,D] flat
{
    const int lane = threadIdx.x & 31;
    const int wave = threadIdx.x >> 5;
    const int b    = blockIdx.y;
    const int qb   = blockIdx.x * 64 + wave * 16;
    const int l15  = lane & 15;
    const int hi   = lane >> 4;

    const _Float16* Kb = Kh + (size_t)b * TT * DD;
    const _Float16* Vb = Vt + (size_t)b * DD * TT;

    // Hoisted Q B-fragments: lane = column q = qb + l15; 16 contiguous halves per chunk.
    const _Float16* qrow = Qh + ((size_t)b * TT + qb + l15) * DD + hi * 16;
    v16h qB[8];
#pragma unroll
    for (int kk = 0; kk < 8; ++kk) qB[kk] = *(const v16h*)(qrow + kk * 32);

    v8f zero = {};
    v8f oacc[16];
#pragma unroll
    for (int dt = 0; dt < 16; ++dt) oacc[dt] = zero;

    float mrun = -INFINITY;   // running max for column q = qb + l15
    float lrun = 0.f;         // running sum for column q

    for (int sb = 0; sb <= qb; sb += 32) {
        // ---- S^T tile 0: keys sb..sb+15 (A-frag = K rows, contiguous) ----
        v8f s0 = zero;
        {
            const _Float16* kr = Kb + (size_t)(sb + l15) * DD + hi * 16;
#pragma unroll
            for (int kk = 0; kk < 8; ++kk) {
                HV ka;
                ka.h[0] = *(const v8h*)(kr + kk * 32);
                ka.h[1] = *(const v8h*)(kr + kk * 32 + 8);
                s0 = __builtin_amdgcn_wmma_f32_16x16x32_f16(false, ka.v, false, qB[kk],
                                                            (short)0, s0, false, false);
            }
        }
        // ---- S^T tile 1: keys sb+16..sb+31 ----
        v8f s1;
        const bool t1_any = (sb + 16 <= qb + 15);
        if (t1_any) {
            s1 = zero;
            const _Float16* kr = Kb + (size_t)(sb + 16 + l15) * DD + hi * 16;
#pragma unroll
            for (int kk = 0; kk < 8; ++kk) {
                HV ka;
                ka.h[0] = *(const v8h*)(kr + kk * 32);
                ka.h[1] = *(const v8h*)(kr + kk * 32 + 8);
                s1 = __builtin_amdgcn_wmma_f32_16x16x32_f16(false, ka.v, false, qB[kk],
                                                            (short)0, s1, false, false);
            }
        } else {
#pragma unroll
            for (int r = 0; r < 8; ++r) s1[r] = -INFINITY;
        }

        // ---- causal mask: element (s = sb + r + hi*8 [+16], q = qb + l15) ----
        const int q = qb + l15;
        if (sb + 15 > qb) {
#pragma unroll
            for (int r = 0; r < 8; ++r)
                if (sb + r + hi * 8 > q) s0[r] = -INFINITY;
        }
        if (t1_any && (sb + 31 > qb)) {
#pragma unroll
            for (int r = 0; r < 8; ++r)
                if (sb + 16 + r + hi * 8 > q) s1[r] = -INFINITY;
        }

        // ---- online softmax: column q stats = slot-reduce + one half-wave swap ----
        float cm = fmaxf(s0[0], s1[0]);
#pragma unroll
        for (int r = 1; r < 8; ++r) cm = fmaxf(cm, fmaxf(s0[r], s1[r]));
        cm = fmaxf(cm, __shfl_xor(cm, 16, 32));

        const float mnew = fmaxf(mrun, cm);
        const float scl  = __expf(mrun - mnew);    // 0 on first chunk
        mrun = mnew;

        v8f p0, p1;
        float cs = 0.f;
#pragma unroll
        for (int r = 0; r < 8; ++r) {
            p0[r] = __expf(s0[r] - mnew);
            p1[r] = __expf(s1[r] - mnew);
            cs += p0[r] + p1[r];
        }
        cs += __shfl_xor(cs, 16, 32);
        lrun = lrun * scl + cs;

        // rescale O^T accumulators (per-lane scalar: column q fixed per lane)
#pragma unroll
        for (int dt = 0; dt < 16; ++dt)
#pragma unroll
            for (int r = 0; r < 8; ++r) oacc[dt][r] *= scl;

        // ---- P^T C-layout -> B-layout: half-wave swap, no LDS ----
        // lane<16 wants [p0(own), p0(lane+16)]; lane>=16 wants [p1(lane-16), p1(own)]
        HV pb;
#pragma unroll
        for (int r = 0; r < 8; ++r) {
            float x0 = __shfl_xor(p0[r], 16, 32);
            float x1 = __shfl_xor(p1[r], 16, 32);
            pb.h[0][r] = (_Float16)(hi ? x1 : p0[r]);
            pb.h[1][r] = (_Float16)(hi ? p1[r] : x0);
        }

        // ---- O^T += V^T @ P^T  (A-frag = V^T rows, contiguous in Vt) ----
#pragma unroll
        for (int dt = 0; dt < 16; ++dt) {
            const _Float16* vr = Vb + (size_t)(dt * 16 + l15) * TT + sb + hi * 16;
            HV va;
            va.h[0] = *(const v8h*)(vr);
            va.h[1] = *(const v8h*)(vr + 8);
            oacc[dt] = __builtin_amdgcn_wmma_f32_16x16x32_f16(false, va.v, false, pb.v,
                                                              (short)0, oacc[dt], false, false);
        }
    }

    // ---- epilogue: each lane owns output row q = qb + l15; d consecutive ----
    const float inv = 1.0f / lrun;
    float* orow = out + ((size_t)b * TT + qb + l15) * DD;
#pragma unroll
    for (int dt = 0; dt < 16; ++dt) {
        v8f o = oacc[dt] * inv;
        *(v8f*)(orow + dt * 16 + hi * 8) = o;
    }
}

extern "C" void kernel_launch(void* const* d_in, const int* in_sizes, int n_in,
                              void* d_out, int out_size, void* d_ws, size_t ws_size,
                              hipStream_t stream) {
    const float* x  = (const float*)d_in[0];
    const float* Wq = (const float*)d_in[1];
    const float* bq = (const float*)d_in[2];
    const float* Wk = (const float*)d_in[3];
    const float* bk = (const float*)d_in[4];
    const float* Wv = (const float*)d_in[5];
    const float* bv = (const float*)d_in[6];
    float* out = (float*)d_out;

    const size_t nX = (size_t)BB * TT * DD;   // 4,194,304
    const size_t nW = (size_t)DD * DD;        // 65,536

    char* ws = (char*)d_ws;
    size_t off = 0;
    _Float16* Xh  = (_Float16*)(ws + off); off += nX * 2;
    _Float16* Qh  = (_Float16*)(ws + off); off += nX * 2;
    _Float16* Kh  = (_Float16*)(ws + off); off += nX * 2;
    _Float16* Vth = (_Float16*)(ws + off); off += nX * 2;
    _Float16* Wqh = (_Float16*)(ws + off); off += nW * 2;
    _Float16* Wkh = (_Float16*)(ws + off); off += nW * 2;
    _Float16* Wvh = (_Float16*)(ws + off); off += nW * 2;

    // convert inputs to f16
    cvt_f32_f16<<<(int)(nX / 256), 256, 0, stream>>>(x,  Xh,  (int)nX);
    cvt_f32_f16<<<(int)(nW / 256), 256, 0, stream>>>(Wq, Wqh, (int)nW);
    cvt_f32_f16<<<(int)(nW / 256), 256, 0, stream>>>(Wk, Wkh, (int)nW);
    cvt_f32_f16<<<(int)(nW / 256), 256, 0, stream>>>(Wv, Wvh, (int)nW);

    // QKV projections (Q pre-scaled by 1/sqrt(D)=1/16; V stored transposed)
    dim3 ggrid(TT / 128, BB);
    qkv_gemm<false><<<ggrid, 256, 0, stream>>>(Xh, Wqh, bq, Qh,  1.0f / 16.0f);
    qkv_gemm<false><<<ggrid, 256, 0, stream>>>(Xh, Wkh, bk, Kh,  1.0f);
    qkv_gemm<true ><<<ggrid, 256, 0, stream>>>(Xh, Wvh, bv, Vth, 1.0f);

    // causal flash attention
    dim3 agrid(TT / 64, BB);
    flash_attn<<<agrid, 128, 0, stream>>>(Qh, Kh, Vth, out);
}
